// GCN_28776280883768
// MI455X (gfx1250) — matrix operand order repaired and checked by reading, not compiled, not run on hardware
//
#include <hip/hip_runtime.h>
#include <hip/hip_bf16.h>

// ---- problem constants (from reference) ----
#define BATCH 16
#define CCH   256
#define HWSZ  4096   // 64*64

typedef __attribute__((ext_vector_type(16))) __bf16 v16bf;
typedef __attribute__((ext_vector_type(8)))  __bf16 v8bf;
typedef __attribute__((ext_vector_type(8)))  float  v8f;

__device__ __forceinline__ unsigned short f32_to_bf16_rne(float f) {
  unsigned int u = __float_as_uint(f);
  unsigned int r = u + 0x7FFFu + ((u >> 16) & 1u);   // round-to-nearest-even
  return (unsigned short)(r >> 16);
}

// ---------------------------------------------------------------------------
// Kernel 0: c_adj[b,c] = mean over HW.  One 256-thread block per (b,c) row.
// ---------------------------------------------------------------------------
__global__ void k_rowmean(const float* __restrict__ x, float* __restrict__ c_adj) {
  const int bc = blockIdx.x;                       // 0 .. B*C-1
  const float4* r4 = (const float4*)(x + (size_t)bc * HWSZ);
  const int t = threadIdx.x;
  float s = 0.f;
#pragma unroll
  for (int i = 0; i < (HWSZ / 4) / 256; ++i) {     // 4 iterations
    float4 v = r4[t + i * 256];
    s += v.x + v.y + v.z + v.w;
  }
  __shared__ float red[256];
  red[t] = s;
  __syncthreads();
  for (int off = 128; off >= 32; off >>= 1) {
    if (t < off) red[t] += red[t + off];
    __syncthreads();
  }
  if (t < 32) {
    float v = red[t];
#pragma unroll
    for (int off = 16; off > 0; off >>= 1) v += __shfl_down(v, off, 32);
    if (t == 0) c_adj[bc] = v * (1.0f / (float)HWSZ);
  }
}

// ---------------------------------------------------------------------------
// Kernel 1: gated[b,i,j] = adj[i,j] * s(b,i,j)  (bf16 out).
// s is symmetric in (i,j) analytically, so the reference's symmetrization
// is the identity and we compute it directly.
// ---------------------------------------------------------------------------
__global__ void k_gate(const float* __restrict__ adj, const float* __restrict__ c_adj,
                       unsigned short* __restrict__ gated) {
  const int idx = blockIdx.x * 256 + threadIdx.x;  // B*C*C total
  const int b  = idx >> 16;                        // / (C*C) == 65536
  const int ij = idx & 65535;
  const int i = ij >> 8, j = ij & 255;
  const float d   = c_adj[b * CCH + j] - c_adj[b * CCH + i];
  const float sig = 1.0f / (1.0f + __expf(-d));
  const float s   = fabsf(fabsf(sig - 0.5f) - 0.5f) * 2.0f;
  gated[idx] = f32_to_bf16_rne(adj[ij] * s);
}

// ---------------------------------------------------------------------------
// Kernel 2: xT[b,n,k] = bf16(x[b,k,n])  (transpose + convert, 32x32 LDS tile)
// Makes the GEMM B-operand K-contiguous so WMMA B fragments are single
// 32-byte contiguous loads per lane.
// ---------------------------------------------------------------------------
__global__ void k_xpose(const float* __restrict__ x, unsigned short* __restrict__ xT) {
  __shared__ unsigned short tile[32][34];          // pad: stride 68B avoids conflicts
  const int b  = blockIdx.z;
  const int c0 = blockIdx.y * 32;
  const int n0 = blockIdx.x * 32;
  const int tx = threadIdx.x & 31;
  const int ty = threadIdx.x >> 5;                 // 0..7
  const float* xb = x + ((size_t)b * CCH + c0) * HWSZ + n0;
#pragma unroll
  for (int i = 0; i < 4; ++i) {
    const int c = ty + 8 * i;
    tile[c][tx] = f32_to_bf16_rne(xb[(size_t)c * HWSZ + tx]);   // coalesced read
  }
  __syncthreads();
  unsigned short* xtb = xT + ((size_t)b * HWSZ + n0) * CCH + c0;
#pragma unroll
  for (int i = 0; i < 4; ++i) {
    const int n = ty + 8 * i;
    xtb[(size_t)n * CCH + tx] = tile[tx][n];                    // coalesced write
  }
}

// ---------------------------------------------------------------------------
// Kernel 3: out[b] = relu( (gated[b] @ fea[b]) * para ), bf16 WMMA, f32 acc.
// Block: 256 threads = 8 wave32, tile 128(M) x 128(N).
// Wave (wm,wn) in 2x4 grid owns 64x32 = 4x2 tiles of 16x16.
// A fragment (16x32 bf16): lane=row(M), lanes 0-15 hold K k..k+15,
//   lanes 16-31 hold K k+16..k+31               -> two 16B loads.
// B fragment (32x16 bf16): lane=col(N), lanes 0-15 hold K k..k+15,
//   lanes 16-31 hold K k+16..k+31               -> one 32B contiguous load.
// ---------------------------------------------------------------------------
__global__ void __launch_bounds__(256)
k_wmma_gemm(const unsigned short* __restrict__ gated_u16,
            const unsigned short* __restrict__ xT_u16,
            const float* __restrict__ para,
            float* __restrict__ out) {
  const int b    = blockIdx.z;
  const int m0   = blockIdx.y * 128;
  const int n0   = blockIdx.x * 128;
  const int wave = threadIdx.x >> 5;
  const int lane = threadIdx.x & 31;
  const int wm    = (wave >> 2) * 64;   // 0 / 64
  const int wn    = (wave & 3) * 32;    // 0 / 32 / 64 / 96
  const int lhalf = lane >> 4;          // 0 / 1
  const int l15   = lane & 15;

  const __bf16* A  = (const __bf16*)gated_u16 + (size_t)b * CCH * CCH;
  const __bf16* Bt = (const __bf16*)xT_u16    + (size_t)b * HWSZ * CCH;

  const v8f vzero = {};
  v8f acc[4][2];
#pragma unroll
  for (int mt = 0; mt < 4; ++mt)
#pragma unroll
    for (int nt = 0; nt < 2; ++nt) acc[mt][nt] = vzero;

  for (int kk = 0; kk < CCH; kk += 32) {
    v16bf af[4];
#pragma unroll
    for (int mt = 0; mt < 4; ++mt) {
      const int row = m0 + wm + mt * 16 + l15;
      const int ka  = kk + lhalf * 16;
      const v8bf lo = *(const v8bf*)(A + (size_t)row * CCH + ka);
      const v8bf hi = *(const v8bf*)(A + (size_t)row * CCH + ka + 8);
#pragma unroll
      for (int e = 0; e < 8; ++e) { af[mt][e] = lo[e]; af[mt][e + 8] = hi[e]; }
    }
    v16bf bfr[2];
#pragma unroll
    for (int nt = 0; nt < 2; ++nt) {
      const int col = n0 + wn + nt * 16 + l15;
      const int kb  = kk + lhalf * 16;
      bfr[nt] = *(const v16bf*)(Bt + (size_t)col * CCH + kb);   // 32B contiguous
    }
#pragma unroll
    for (int mt = 0; mt < 4; ++mt)
#pragma unroll
      for (int nt = 0; nt < 2; ++nt)
        acc[mt][nt] = __builtin_amdgcn_wmma_f32_16x16x32_bf16(
            false, af[mt], false, bfr[nt], (short)0, acc[mt][nt], false, false);
  }

  // Epilogue: C/D layout -> VGPR r holds (row r | r+8 per lane half), col = lane&15
#pragma unroll
  for (int mt = 0; mt < 4; ++mt) {
#pragma unroll
    for (int nt = 0; nt < 2; ++nt) {
      const int colg = n0 + wn + nt * 16 + l15;
#pragma unroll
      for (int r = 0; r < 8; ++r) {
        const int rowg = m0 + wm + mt * 16 + r + lhalf * 8;
        const float v = acc[mt][nt][r] * para[(size_t)rowg * HWSZ + colg];
        out[((size_t)b * CCH + rowg) * HWSZ + colg] = fmaxf(v, 0.0f);
      }
    }
  }
}

// ---------------------------------------------------------------------------
extern "C" void kernel_launch(void* const* d_in, const int* in_sizes, int n_in,
                              void* d_out, int out_size, void* d_ws, size_t ws_size,
                              hipStream_t stream) {
  const float* x    = (const float*)d_in[0];   // (B,C,H,W) f32
  const float* adj  = (const float*)d_in[1];   // (C,C)     f32
  const float* para = (const float*)d_in[2];   // (1,C,H,W) f32
  float* out        = (float*)d_out;           // (B,C,H,W) f32

  // workspace layout (256B-aligned offsets):
  //   [0,16K)        c_adj  f32  (B*C)
  //   [16K,16K+2M)   gated  bf16 (B*C*C)
  //   [16K+2M, +32M) xT     bf16 (B*HW*C)
  float*          c_adj = (float*)d_ws;
  unsigned short* gated = (unsigned short*)((char*)d_ws + (16u << 10));
  unsigned short* xT    = (unsigned short*)((char*)d_ws + (16u << 10) + (2u << 20));

  k_rowmean<<<dim3(BATCH * CCH), dim3(256), 0, stream>>>(x, c_adj);
  k_gate<<<dim3((BATCH * CCH * CCH) / 256), dim3(256), 0, stream>>>(adj, c_adj, gated);
  k_xpose<<<dim3(HWSZ / 32, CCH / 32, BATCH), dim3(256), 0, stream>>>(x, xT);
  k_wmma_gemm<<<dim3(HWSZ / 128, CCH / 128, BATCH), dim3(256), 0, stream>>>(gated, xT, para, out);
}
